// EGCN2_13975823581725
// MI455X (gfx1250) — matrix operand order repaired and checked by reading, not compiled, not run on hardware
//
#include <hip/hip_runtime.h>
#include <hip/hip_bf16.h>
#include <math.h>
#include <stdint.h>

typedef __attribute__((ext_vector_type(16))) _Float16 v16h;
typedef __attribute__((ext_vector_type(8)))  float    v8f;

#define NNODES 25000
#define NEDGES 250000
#define NPAD   25008          // 16 * 1563
#define HDIM   128

__device__ __forceinline__ float siluf(float v) { return v / (1.0f + __expf(-v)); }
__device__ __forceinline__ float lreluf(float v) { return v > 0.0f ? v : 0.01f * v; }

__device__ __forceinline__ v8f wmma_f16(v16h a, v16h b, v8f c) {
    // D = A(16x32 f16) * B(32x16 f16) + C(16x16 f32)
    return __builtin_amdgcn_wmma_f32_16x16x32_f16(false, a, false, b, (short)0, c,
                                                  false, false);
}

// Per-wave LDS RAW fence: DS unit is in-order per wave; this stops compiler
// reordering and drains DScnt before the dependent ds_loads.
__device__ __forceinline__ void dsfence() {
    asm volatile("s_wait_dscnt 0x0" ::: "memory");
}

// Async global -> LDS staging (CDNA5 ASYNCcnt path), 16B per lane per call.
__device__ __forceinline__ void async_copy16(uint32_t lds_addr, const void* gptr) {
    asm volatile("global_load_async_to_lds_b128 %0, %1, off"
                 :: "v"(lds_addr), "v"((uint64_t)(uintptr_t)gptr)
                 : "memory");
}
__device__ __forceinline__ void async_wait0() {
    asm volatile("s_wait_asynccnt 0x0" ::: "memory");
}

// ---- fragment loaders ---------------------------------------------------
// B fragment: packed as [frag][lane][16 halves], frag = kc*8 + nc
__device__ __forceinline__ v16h load_bfrag(const _Float16* p, int frag, int lane) {
    const _Float16* q = p + (size_t)frag * 512 + lane * 16;
    v16h b;
    *(uint4*)&b         = *(const uint4*)q;        // halves 0..7
    *(((uint4*)&b) + 1) = *(const uint4*)(q + 8);  // halves 8..15
    return b;
}

// A fragment from an f16 row (row-major features). 16-bit A 16x32 layout:
// lane half hp (=l>>4): v0..3 hold k = kb + hp*8 + {0..7}, v4..7 hold +16.
__device__ __forceinline__ v16h load_afrag_h(const _Float16* row, int kb, int hp) {
    const _Float16* p = row + kb + hp * 8;
    v16h a;
    *(uint4*)&a         = *(const uint4*)p;
    *(((uint4*)&a) + 1) = *(const uint4*)(p + 16);
    return a;
}

// A fragment from an f32 row (h_neigh accumulator), converted to f16.
__device__ __forceinline__ v16h load_afrag_f(const float* row, int kb, int hp) {
    const float* p = row + kb + hp * 8;
    v16h a;
    _Float16* ah = (_Float16*)&a;
#pragma unroll
    for (int i = 0; i < 8; ++i) {
        ah[i]     = (_Float16)p[i];
        ah[8 + i] = (_Float16)p[16 + i];
    }
    return a;
}

// One K-step of a 16xK @ Kx128 GEMM: 8 B fragments in distinct registers,
// interleaved with the 8 accumulating WMMAs so ds_loads pipeline with the
// matrix unit instead of draining DScnt per fragment.
__device__ __forceinline__ void wmma_kstep(v8f* acc, v16h a, const _Float16* B,
                                           int kc, int l) {
    v16h b0 = load_bfrag(B, kc * 8 + 0, l);
    v16h b1 = load_bfrag(B, kc * 8 + 1, l);
    v16h b2 = load_bfrag(B, kc * 8 + 2, l);
    v16h b3 = load_bfrag(B, kc * 8 + 3, l);
    acc[0] = wmma_f16(a, b0, acc[0]);
    acc[1] = wmma_f16(a, b1, acc[1]);
    v16h b4 = load_bfrag(B, kc * 8 + 4, l);
    v16h b5 = load_bfrag(B, kc * 8 + 5, l);
    acc[2] = wmma_f16(a, b2, acc[2]);
    acc[3] = wmma_f16(a, b3, acc[3]);
    v16h b6 = load_bfrag(B, kc * 8 + 6, l);
    v16h b7 = load_bfrag(B, kc * 8 + 7, l);
    acc[4] = wmma_f16(a, b4, acc[4]);
    acc[5] = wmma_f16(a, b5, acc[5]);
    acc[6] = wmma_f16(a, b6, acc[6]);
    acc[7] = wmma_f16(a, b7, acc[7]);
}

// ---- weight packer ------------------------------------------------------
// W is [K][128] row-major f32. Output B fragments for 16x16x32 f16 WMMA.
// grid.x = (Kpad/32)*8 fragments, block = 32 lanes. k >= K rows are zeroed.
__global__ void pack_b_kernel(const float* __restrict__ W, _Float16* __restrict__ out,
                              int K) {
    int frag = blockIdx.x;
    int kc = frag >> 3, nc = frag & 7;
    int l = threadIdx.x;
#pragma unroll
    for (int v = 0; v < 8; ++v) {
#pragma unroll
        for (int hh = 0; hh < 2; ++hh) {
            int k = kc * 32 + (v >> 2) * 16 + (l >> 4) * 8 + (v & 3) * 2 + hh;
            int n = nc * 16 + (l & 15);
            float val = (k < K) ? W[(size_t)k * 128 + n] : 0.0f;
            out[(size_t)frag * 512 + l * 16 + v * 2 + hh] = (_Float16)val;
        }
    }
}

// ---- init / zero kernels ------------------------------------------------
__global__ void init_nf16_kernel(const float* __restrict__ nf, _Float16* __restrict__ out) {
    size_t idx = (size_t)blockIdx.x * 256 + threadIdx.x;
    if (idx >= (size_t)NPAD * 32) return;
    size_t r = idx / 32;
    out[idx] = (r < NNODES) ? (_Float16)nf[idx] : (_Float16)0.0f;
}

__global__ void init_x_kernel(const float* __restrict__ cf, float* __restrict__ x) {
    size_t idx = (size_t)blockIdx.x * 256 + threadIdx.x;
    if (idx >= (size_t)NPAD * 2) return;
    size_t r = idx / 2;
    x[idx] = (r < NNODES) ? cf[idx] : 0.0f;
}

__global__ void zero_f_kernel(float* __restrict__ p, size_t n) {
    size_t idx = (size_t)blockIdx.x * 256 + threadIdx.x;
    if (idx < n) p[idx] = 0.0f;
}

// ---- edge kernel: 4 waves/block, weights staged in LDS ------------------
// f = [h_src | h_dst | radial]  (K = 2*32*SC + 1, padded to K1C*32)
// msg_h = SiLU(SiLU(f@We1+be1)@We2+be2); coef = SiLU(msg_h@Wc1+bc1)@Wc2
// atomics: h_neigh[dst] += msg_h ; x_sum[dst] += coef*x_hat ; deg[dst] += 1
template <int SC>
__global__ __launch_bounds__(128) void egnn_edge_kernel(
    const _Float16* hin16, int hstride, const float* __restrict__ x,
    const int* __restrict__ src, const int* __restrict__ dst,
    const _Float16* __restrict__ Wpack,  // We1p | We2p | Wc1p contiguous
    const float* __restrict__ be1, const float* __restrict__ be2,
    const float* __restrict__ bc1, const float* __restrict__ Wc2,
    float* __restrict__ h_neigh, float* __restrict__ x_sum,
    float* __restrict__ deg, int ntiles) {
    constexpr int K1C = 2 * SC + 1;
    constexpr int FR1 = K1C * 8;       // We1 fragments
    constexpr int WFR = FR1 + 64;      // + We2 (32) + Wc1 (32)
    const int STRH = 136;              // halves; 272B rows -> 16B aligned segs

    __shared__ _Float16 wlds[WFR * 512];     // staged B fragments
    __shared__ _Float16 tlds[4][16 * 136];   // per-wave transpose tiles

    // --- cooperative async staging of all B fragments ---
    {
        uint32_t lbase = (uint32_t)(uintptr_t)&wlds[0];
        for (int i = threadIdx.x; i < WFR * 64; i += 128)
            async_copy16(lbase + i * 16, Wpack + (size_t)i * 8);
        async_wait0();
    }
    __syncthreads();

    const _Float16* B1 = wlds;                       // We1p
    const _Float16* B2 = wlds + (size_t)FR1 * 512;   // We2p
    const _Float16* B3 = B2 + (size_t)32 * 512;      // Wc1p

    int l = threadIdx.x & 31;
    int w = threadIdx.x >> 5;
    int m = l & 15, hp = l >> 4;
    _Float16* mytile = tlds[w];

    for (int t = blockIdx.x * 4 + w; t < ntiles; t += gridDim.x * 4) {
        size_t e = (size_t)t * 16 + m;
        int s = src[e], d = dst[e];
        float2 xs = ((const float2*)x)[s];
        float2 xd = ((const float2*)x)[d];
        float dx0 = xs.x - xd.x, dx1 = xs.y - xd.y;
        float radial = dx0 * dx0 + dx1 * dx1;
        float rinv = 1.0f / (sqrtf(radial) + 1e-30f);
        dx0 *= rinv;
        dx1 *= rinv;

        const _Float16* rs = hin16 + (size_t)s * hstride;
        const _Float16* rd = hin16 + (size_t)d * hstride;

        // dst row per D-fragment row m' = v + 8*hp
        int drow[8];
#pragma unroll
        for (int v = 0; v < 8; ++v) drow[v] = __shfl(d, v + 8 * hp, 32);

        v8f zero = {};
        v8f acc[8];

        // ---- GEMM1: f @ We1 ----
#pragma unroll
        for (int nc = 0; nc < 8; ++nc) acc[nc] = zero;
#pragma unroll
        for (int kc = 0; kc < K1C; ++kc) {
            v16h a;
            if (kc < SC) {
                a = load_afrag_h(rs, kc * 32, hp);
            } else if (kc < 2 * SC) {
                a = load_afrag_h(rd, (kc - SC) * 32, hp);
            } else {  // radial column (local k 0 of last chunk)
                v16h z = {};
                a = z;
                _Float16* ah = (_Float16*)&a;
                if (hp == 0) ah[0] = (_Float16)radial;
            }
            wmma_kstep(acc, a, B1, kc, l);
        }
        // bias + SiLU -> LDS f16 tile t1[16][128]
#pragma unroll
        for (int nc = 0; nc < 8; ++nc) {
            float b = be1[nc * 16 + m];
#pragma unroll
            for (int v = 0; v < 8; ++v)
                mytile[(v + 8 * hp) * STRH + nc * 16 + m] =
                    (_Float16)siluf(acc[nc][v] + b);
        }
        dsfence();

        // ---- GEMM2: t1 @ We2 ----
#pragma unroll
        for (int nc = 0; nc < 8; ++nc) acc[nc] = zero;
#pragma unroll
        for (int kc = 0; kc < 4; ++kc) {
            v16h a = load_afrag_h(mytile + m * STRH, kc * 32, hp);
            wmma_kstep(acc, a, B2, kc, l);
        }
        dsfence();
        // bias + SiLU = msg_h: scatter-add into h_neigh + restage into LDS
#pragma unroll
        for (int nc = 0; nc < 8; ++nc) {
            float b = be2[nc * 16 + m];
            int n = nc * 16 + m;
#pragma unroll
            for (int v = 0; v < 8; ++v) {
                float val = siluf(acc[nc][v] + b);
                atomicAdd(&h_neigh[(size_t)drow[v] * HDIM + n], val);
                mytile[(v + 8 * hp) * STRH + n] = (_Float16)val;
            }
        }
        dsfence();

        // ---- GEMM3: msg_h @ Wc1 ----
#pragma unroll
        for (int nc = 0; nc < 8; ++nc) acc[nc] = zero;
#pragma unroll
        for (int kc = 0; kc < 4; ++kc) {
            v16h a = load_afrag_h(mytile + m * STRH, kc * 32, hp);
            wmma_kstep(acc, a, B3, kc, l);
        }
        dsfence();
        // bias + SiLU -> LDS f16 tile t3
#pragma unroll
        for (int nc = 0; nc < 8; ++nc) {
            float b = bc1[nc * 16 + m];
#pragma unroll
            for (int v = 0; v < 8; ++v)
                mytile[(v + 8 * hp) * STRH + nc * 16 + m] =
                    (_Float16)siluf(acc[nc][v] + b);
        }
        dsfence();

        // coef = t3 @ Wc2 (lane half hp does 64 elems, combine via shfl)
        const _Float16* trow = mytile + m * STRH + hp * 64;
        float sum = 0.0f;
#pragma unroll
        for (int k = 0; k < 64; ++k) sum += (float)trow[k] * Wc2[hp * 64 + k];
        sum += __shfl_xor(sum, 16, 32);

        if (hp == 0) {
            atomicAdd(&x_sum[2 * (size_t)d + 0], sum * dx0);
            atomicAdd(&x_sum[2 * (size_t)d + 1], sum * dx1);
            atomicAdd(&deg[d], 1.0f);
        }
        dsfence();  // tile reuse across loop iterations
    }
}

// ---- node kernel: 4 waves/block, weights staged in LDS ------------------
// h_out = lrelu(SiLU(cat(h, h_neigh)@Wn1+bn1)@Wn2+bn2)
// x     = lrelu(x + x_sum/max(deg,1))
template <int SC>
__global__ __launch_bounds__(128) void egnn_node_kernel(
    const _Float16* hin16, int hstride, const float* __restrict__ h_neigh,
    const _Float16* __restrict__ Wpack,  // Wn1p | Wn2p contiguous
    const float* __restrict__ bn1, const float* __restrict__ bn2,
    const float* __restrict__ x_sum, const float* __restrict__ deg, float* x,
    _Float16* hout16, float* __restrict__ houtf, int ntiles) {
    constexpr int KC = SC + 4;
    constexpr int FR1 = KC * 8;
    constexpr int WFR = FR1 + 32;
    const int STRH = 136;

    __shared__ _Float16 wlds[WFR * 512];
    __shared__ _Float16 tlds[4][16 * 136];

    {
        uint32_t lbase = (uint32_t)(uintptr_t)&wlds[0];
        for (int i = threadIdx.x; i < WFR * 64; i += 128)
            async_copy16(lbase + i * 16, Wpack + (size_t)i * 8);
        async_wait0();
    }
    __syncthreads();

    const _Float16* B1 = wlds;
    const _Float16* B2 = wlds + (size_t)FR1 * 512;

    int l = threadIdx.x & 31;
    int w = threadIdx.x >> 5;
    int m = l & 15, hp = l >> 4;
    _Float16* mytile = tlds[w];

    for (int t = blockIdx.x * 4 + w; t < ntiles; t += gridDim.x * 4) {
        size_t base = (size_t)t * 16;
        size_t i = base + m;
        const _Float16* rh = hin16 + i * hstride;
        const float* rn = h_neigh + i * HDIM;

        v8f zero = {};
        v8f acc[8];
#pragma unroll
        for (int nc = 0; nc < 8; ++nc) acc[nc] = zero;
#pragma unroll
        for (int kc = 0; kc < KC; ++kc) {
            v16h a = (kc < SC) ? load_afrag_h(rh, kc * 32, hp)
                               : load_afrag_f(rn, (kc - SC) * 32, hp);
            wmma_kstep(acc, a, B1, kc, l);
        }
#pragma unroll
        for (int nc = 0; nc < 8; ++nc) {
            float b = bn1[nc * 16 + m];
#pragma unroll
            for (int v = 0; v < 8; ++v)
                mytile[(v + 8 * hp) * STRH + nc * 16 + m] =
                    (_Float16)siluf(acc[nc][v] + b);
        }
        dsfence();

#pragma unroll
        for (int nc = 0; nc < 8; ++nc) acc[nc] = zero;
#pragma unroll
        for (int kc = 0; kc < 4; ++kc) {
            v16h a = load_afrag_h(mytile + m * STRH, kc * 32, hp);
            wmma_kstep(acc, a, B2, kc, l);
        }
        // bn2 + post-layer LeakyReLU; write h (f32 + f16)
#pragma unroll
        for (int nc = 0; nc < 8; ++nc) {
            float b = bn2[nc * 16 + m];
            int n = nc * 16 + m;
#pragma unroll
            for (int v = 0; v < 8; ++v) {
                size_t r = base + v + 8 * hp;
                float val = lreluf(acc[nc][v] + b);
                houtf[r * HDIM + n] = val;
                hout16[r * HDIM + n] = (_Float16)val;
            }
        }
        // coordinate update: lane handles node (base+m), component hp
        float dg = fmaxf(deg[i], 1.0f);
        float xv = x[2 * i + hp] + x_sum[2 * i + hp] / dg;
        x[2 * i + hp] = lreluf(xv);
        dsfence();  // tile reuse across loop iterations
    }
}

// ---- classifier head ----------------------------------------------------
__global__ void head_kernel(const float* __restrict__ hf, const float* __restrict__ Wh,
                            const float* __restrict__ bh, float* __restrict__ out) {
    int n = blockIdx.x * blockDim.x + threadIdx.x;
    if (n >= NNODES) return;
    const float* r = hf + (size_t)n * HDIM;
    float a0 = bh[0], a1 = bh[1], a2 = bh[2];
#pragma unroll 8
    for (int k = 0; k < HDIM; ++k) {
        float hv = r[k];
        a0 += hv * Wh[k * 3 + 0];
        a1 += hv * Wh[k * 3 + 1];
        a2 += hv * Wh[k * 3 + 2];
    }
    out[n * 3 + 0] = a0;
    out[n * 3 + 1] = a1;
    out[n * 3 + 2] = a2;
}

// ---- host launch --------------------------------------------------------
extern "C" void kernel_launch(void* const* d_in, const int* in_sizes, int n_in,
                              void* d_out, int out_size, void* d_ws, size_t ws_size,
                              hipStream_t stream) {
    (void)in_sizes; (void)n_in; (void)out_size; (void)ws_size;
    const float* node_feat  = (const float*)d_in[0];
    const float* coord_feat = (const float*)d_in[1];
    const int*   src        = (const int*)d_in[2];
    const int*   dst        = (const int*)d_in[3];
    const float* l1_We1 = (const float*)d_in[4];
    const float* l1_be1 = (const float*)d_in[5];
    const float* l1_We2 = (const float*)d_in[6];
    const float* l1_be2 = (const float*)d_in[7];
    const float* l1_Wn1 = (const float*)d_in[8];
    const float* l1_bn1 = (const float*)d_in[9];
    const float* l1_Wn2 = (const float*)d_in[10];
    const float* l1_bn2 = (const float*)d_in[11];
    const float* l1_Wc1 = (const float*)d_in[12];
    const float* l1_bc1 = (const float*)d_in[13];
    const float* l1_Wc2 = (const float*)d_in[14];
    const float* We1 = (const float*)d_in[15];
    const float* be1 = (const float*)d_in[16];
    const float* We2 = (const float*)d_in[17];
    const float* be2 = (const float*)d_in[18];
    const float* Wn1 = (const float*)d_in[19];
    const float* bn1 = (const float*)d_in[20];
    const float* Wn2 = (const float*)d_in[21];
    const float* bn2 = (const float*)d_in[22];
    const float* Wc1 = (const float*)d_in[23];
    const float* bc1 = (const float*)d_in[24];
    const float* Wc2 = (const float*)d_in[25];
    const float* Wh  = (const float*)d_in[26];
    const float* bh  = (const float*)d_in[27];

    // ---- workspace carve ----
    char* base = (char*)d_ws;
    size_t off = 0;
    auto carve = [&](size_t bytes) -> void* {
        off = (off + 255) & ~(size_t)255;
        void* p = base + off;
        off += bytes;
        return p;
    };
    // contiguous atomic accumulators (zeroed together): h_neigh | x_sum | deg
    float* zblk   = (float*)carve((size_t)NPAD * 131 * sizeof(float));
    float* hneigh = zblk;
    float* xsum   = zblk + (size_t)NPAD * 128;
    float* degp   = xsum + (size_t)NPAD * 2;
    float*     xbuf = (float*)carve((size_t)NPAD * 2 * sizeof(float));
    _Float16*  h16  = (_Float16*)carve((size_t)NPAD * 128 * sizeof(_Float16));
    _Float16*  nf16 = (_Float16*)carve((size_t)NPAD * 32 * sizeof(_Float16));
    float*     hf32 = (float*)carve((size_t)NPAD * 128 * sizeof(float));
    _Float16*  pk   = (_Float16*)carve((size_t)557056 * sizeof(_Float16));

    // packed weight sub-offsets (in halves); each layer keeps
    // We1|We2|Wc1 contiguous (edge staging) and Wn1|Wn2 contiguous (node).
    _Float16* p_l1_We1 = pk;                    // 96x128
    _Float16* p_l1_We2 = p_l1_We1 + 12288;      // 128x128
    _Float16* p_l1_Wc1 = p_l1_We2 + 16384;      // 128x128
    _Float16* p_l1_Wn1 = p_l1_Wc1 + 16384;      // 160x128
    _Float16* p_l1_Wn2 = p_l1_Wn1 + 20480;      // 128x128
    _Float16* pL       = p_l1_Wn2 + 16384;      // 4 x layer block (118784 each)
    const size_t oWe1 = 0, oWe2 = 36864, oWc1 = 53248, oWn1 = 69632, oWn2 = 102400;

    // ---- pack weights into WMMA-B fragment layout ----
    pack_b_kernel<<<(96 / 32) * 8, 32, 0, stream>>>(l1_We1, p_l1_We1, 65);
    pack_b_kernel<<<32, 32, 0, stream>>>(l1_We2, p_l1_We2, 128);
    pack_b_kernel<<<32, 32, 0, stream>>>(l1_Wc1, p_l1_Wc1, 128);
    pack_b_kernel<<<(160 / 32) * 8, 32, 0, stream>>>(l1_Wn1, p_l1_Wn1, 160);
    pack_b_kernel<<<32, 32, 0, stream>>>(l1_Wn2, p_l1_Wn2, 128);
    for (int i = 0; i < 4; ++i) {
        _Float16* pw = pL + (size_t)i * 118784;
        pack_b_kernel<<<(288 / 32) * 8, 32, 0, stream>>>(We1 + (size_t)i * 257 * 128, pw + oWe1, 257);
        pack_b_kernel<<<32, 32, 0, stream>>>(We2 + (size_t)i * 16384, pw + oWe2, 128);
        pack_b_kernel<<<32, 32, 0, stream>>>(Wc1 + (size_t)i * 16384, pw + oWc1, 128);
        pack_b_kernel<<<(256 / 32) * 8, 32, 0, stream>>>(Wn1 + (size_t)i * 256 * 128, pw + oWn1, 256);
        pack_b_kernel<<<32, 32, 0, stream>>>(Wn2 + (size_t)i * 16384, pw + oWn2, 128);
    }

    // ---- init node features (f16, padded) and coordinates ----
    init_nf16_kernel<<<((size_t)NPAD * 32 + 255) / 256, 256, 0, stream>>>(node_feat, nf16);
    init_x_kernel<<<((size_t)NPAD * 2 + 255) / 256, 256, 0, stream>>>(coord_feat, xbuf);

    const size_t zn = (size_t)NPAD * 131;
    const int ET = NEDGES / 16;  // 15625 edge tiles
    const int NT = NPAD / 16;    // 1563 node tiles
    const int EG = 512;          // edge grid (4 waves/block)
    const int NG = 256;          // node grid

    // ---- layer 1 (IN_F = 32 -> SC = 1) ----
    zero_f_kernel<<<(zn + 255) / 256, 256, 0, stream>>>(zblk, zn);
    egnn_edge_kernel<1><<<EG, 128, 0, stream>>>(
        nf16, 32, xbuf, src, dst, p_l1_We1, l1_be1, l1_be2, l1_bc1, l1_Wc2,
        hneigh, xsum, degp, ET);
    egnn_node_kernel<1><<<NG, 128, 0, stream>>>(
        nf16, 32, hneigh, p_l1_Wn1, l1_bn1, l1_bn2, xsum, degp, xbuf, h16, hf32,
        NT);

    // ---- layers 2..5 (H = 128 -> SC = 4) ----
    for (int i = 0; i < 4; ++i) {
        _Float16* pw = pL + (size_t)i * 118784;
        zero_f_kernel<<<(zn + 255) / 256, 256, 0, stream>>>(zblk, zn);
        egnn_edge_kernel<4><<<EG, 128, 0, stream>>>(
            h16, 128, xbuf, src, dst, pw + oWe1, be1 + (size_t)i * 128,
            be2 + (size_t)i * 128, bc1 + (size_t)i * 128, Wc2 + (size_t)i * 128,
            hneigh, xsum, degp, ET);
        egnn_node_kernel<4><<<NG, 128, 0, stream>>>(
            h16, 128, hneigh, pw + oWn1, bn1 + (size_t)i * 128,
            bn2 + (size_t)i * 128, xsum, degp, xbuf, h16, hf32, NT);
    }

    // ---- classifier head ----
    head_kernel<<<(NNODES + 127) / 128, 128, 0, stream>>>(hf32, Wh, bh, (float*)d_out);
}
